// FCGATEncoder_76192719831646
// MI455X (gfx1250) — matrix operand order repaired and checked by reading, not compiled, not run on hardware
//
#include <hip/hip_runtime.h>
#include <hip/hip_bf16.h>

// ---------------------------------------------------------------------------
// FCGAT encoder for MI455X (gfx1250, wave32, WMMA).
// GEMMs: v_wmma_f32_16x16x32_bf16, LDS tiles stored in fragment order so
// fragment loads are ds_load_b128 pairs and staging uses b128/b64 stores.
// ---------------------------------------------------------------------------

typedef __bf16 bf16_t;
typedef __attribute__((ext_vector_type(16))) __bf16 v16bf;
typedef __attribute__((ext_vector_type(8)))  float  v8f;

#define Dc     256
#define D3c    768
#define Nn     128
#define Bb     2
#define Hh     4
#define DHh    64
#define BNc    (Bb * Nn)            // 256
#define BNNc   (Bb * Nn * Nn)       // 32768
#define CHUNKR 8192                 // edge-update row chunk
#define NCHUNK 4
#define NEG_INF_F (-4294967295.0f)

__device__ __forceinline__ unsigned pk2(float lo, float hi) {
    unsigned a = __builtin_bit_cast(unsigned, lo) >> 16;
    unsigned b = __builtin_bit_cast(unsigned, hi) & 0xffff0000u;
    return b | a;                     // packed bf16 pair (truncate)
}

// ---------------------------------------------------------------------------
// WMMA GEMM: C[M,Nc] = A[M,K] @ W[K,Nc] (+bias[Nc]) (+resid[M,Nc]) (relu)
// Block: 128 threads = 4 waves. Block tile 64(M) x 64(N); K step 32.
// Wave w owns M sub-tile w and iterates 4 N sub-tiles (4 WMMA / K-step).
// LDS tiles are kept in *fragment order*: [tile][lane][h], 16 bf16 per lane
// contiguous -> fragment load = 2x ds_load_b128.
// Requires: M % 64 == 0, Nc % 64 == 0, K % 32 == 0.
// ---------------------------------------------------------------------------
__global__ __launch_bounds__(128)
void gemm_bf16_wmma(const float* __restrict__ A, const float* __restrict__ W,
                    const float* __restrict__ bias, const float* __restrict__ resid,
                    float* __restrict__ C, int M, int K, int Nc, int relu)
{
    __shared__ __align__(32) unsigned short aF[4][32][16];  // [Mtile][lane][h]
    __shared__ __align__(32) unsigned short bF[4][32][16];  // [Ntile][lane][h]

    const int tid  = threadIdx.x;
    const int wave = tid >> 5;      // 0..3  (M sub-tile)
    const int lane = tid & 31;
    const int m0 = blockIdx.x * 64;
    const int n0 = blockIdx.y * 64;

    v8f acc0 = {}, acc1 = {}, acc2 = {}, acc3 = {};

    for (int k0 = 0; k0 < K; k0 += 32) {
        // ---- stage A tile 64x32 in fragment order: 256 units of 8 consecutive k
        #pragma unroll
        for (int t = 0; t < 2; ++t) {
            int u  = tid + t * 128;          // 0..255
            int m  = u >> 2;                 // 0..63
            int kg = u & 3;                  // 0..3 (8 k's each)
            const float* src = A + (size_t)(m0 + m) * K + k0 + kg * 8;
            float4 f0 = *(const float4*)(src);
            float4 f1 = *(const float4*)(src + 4);
            uint4 pkv;
            pkv.x = pk2(f0.x, f0.y); pkv.y = pk2(f0.z, f0.w);
            pkv.z = pk2(f1.x, f1.y); pkv.w = pk2(f1.z, f1.w);
            int g  = kg & 1;                 // lane group
            int hb = (kg >> 1) << 3;         // h base (0 or 8)
            *(uint4*)&aF[m >> 4][g * 16 + (m & 15)][hb] = pkv;
        }
        // ---- stage B tile 32x64 in fragment order: 512 units of 4 consecutive k
        #pragma unroll
        for (int t = 0; t < 4; ++t) {
            int u  = tid + t * 128;          // 0..511
            int nt = u >> 7;                 // N sub-tile
            int r  = u & 127;
            int n  = r & 15;
            int q  = r >> 4;                 // 0..7 -> k = q*4
            const float* src = W + (size_t)(k0 + q * 4) * Nc + n0 + nt * 16 + n;
            float x0 = src[0];
            float x1 = src[(size_t)Nc];
            float x2 = src[2 * (size_t)Nc];
            float x3 = src[3 * (size_t)Nc];
            unsigned long long p = (unsigned long long)pk2(x0, x1)
                                 | ((unsigned long long)pk2(x2, x3) << 32);
            int g  = (q >> 1) & 1;           // lane group
            int hb = ((q >> 2) << 3) + (q & 1) * 4;  // h base
            *(unsigned long long*)&bF[nt][g * 16 + n][hb] = p;
        }
        if (k0 + 32 < K)
            __builtin_prefetch(A + (size_t)(m0 + (tid >> 1)) * K + (k0 + 32), 0, 1);
        __syncthreads();

        // ---- fragments: contiguous 32B per lane
        v16bf afrag = *(const v16bf*)&aF[wave][lane][0];
        v16bf b0 = *(const v16bf*)&bF[0][lane][0];
        v16bf b1 = *(const v16bf*)&bF[1][lane][0];
        v16bf b2 = *(const v16bf*)&bF[2][lane][0];
        v16bf b3 = *(const v16bf*)&bF[3][lane][0];
        acc0 = __builtin_amdgcn_wmma_f32_16x16x32_bf16(false, afrag, false, b0, (short)0, acc0, false, false);
        acc1 = __builtin_amdgcn_wmma_f32_16x16x32_bf16(false, afrag, false, b1, (short)0, acc1, false, false);
        acc2 = __builtin_amdgcn_wmma_f32_16x16x32_bf16(false, afrag, false, b2, (short)0, acc2, false, false);
        acc3 = __builtin_amdgcn_wmma_f32_16x16x32_bf16(false, afrag, false, b3, (short)0, acc3, false, false);
        __syncthreads();
    }

    // ---- epilogue. C layout: VGPR r, lane l: M = r + (l>=16 ? 8:0), N = l&15
    const int mbase = m0 + wave * 16 + ((lane >> 4) << 3);
    const int nc0   = n0 + (lane & 15);
    v8f* accs[4] = { &acc0, &acc1, &acc2, &acc3 };
    #pragma unroll
    for (int nt = 0; nt < 4; ++nt) {
        int n = nc0 + nt * 16;
        float bv = bias ? bias[n] : 0.f;
        #pragma unroll
        for (int r = 0; r < 8; ++r) {
            int m = mbase + r;
            float v = (*accs[nt])[r] + bv;
            if (resid) v += resid[(size_t)m * Nc + n];
            if (relu)  v = v > 0.f ? v : 0.f;
            C[(size_t)m * Nc + n] = v;
        }
    }
}

// ---------------------------------------------------------------------------
// w_eatt[d][h] = sum_k W_E[d, h*64+k] * a_edge[h,k]   (launch <<<1,256>>>)
// ---------------------------------------------------------------------------
__global__ __launch_bounds__(256)
void weatt_kernel(const float* __restrict__ WE, const float* __restrict__ aedge,
                  float* __restrict__ weatt)
{
    int d = threadIdx.x;
    #pragma unroll
    for (int h = 0; h < Hh; ++h) {
        float s = 0.f;
        #pragma unroll 8
        for (int k = 0; k < DHh; ++k)
            s += WE[(size_t)d * Dc + h * DHh + k] * aedge[h * DHh + k];
        weatt[d * Hh + h] = s;
    }
}

// s_att[b,n,h], d_att[b,n,h] from xp  (1024 threads total)
__global__ __launch_bounds__(256)
void satt_kernel(const float* __restrict__ xp, const float* __restrict__ asrc,
                 const float* __restrict__ adst, float* __restrict__ sA,
                 float* __restrict__ dA)
{
    int idx = blockIdx.x * 256 + threadIdx.x;
    if (idx >= BNc * Hh) return;
    int h = idx & 3, bn = idx >> 2;
    const float* x = xp + (size_t)bn * Dc + h * DHh;
    float s = 0.f, d = 0.f;
    #pragma unroll 8
    for (int k = 0; k < DHh; ++k) { s += x[k] * asrc[h * DHh + k]; d += x[k] * adst[h * DHh + k]; }
    sA[idx] = s; dA[idx] = d;
}

// e_att[row,h] = edge[row,:] @ w_eatt[:,h]   (one thread per edge row)
__global__ __launch_bounds__(256)
void eatt_kernel(const float* __restrict__ edge, const float* __restrict__ weatt,
                 float* __restrict__ eA)
{
    size_t row = (size_t)blockIdx.x * 256 + threadIdx.x;
    const float* e = edge + row * Dc;
    float a0 = 0, a1 = 0, a2 = 0, a3 = 0;
    for (int d = 0; d < Dc; ++d) {
        float v = e[d];
        a0 += v * weatt[d * 4 + 0];
        a1 += v * weatt[d * 4 + 1];
        a2 += v * weatt[d * 4 + 2];
        a3 += v * weatt[d * 4 + 3];
    }
    eA[row * 4 + 0] = a0; eA[row * 4 + 1] = a1;
    eA[row * 4 + 2] = a2; eA[row * 4 + 3] = a3;
}

// ---------------------------------------------------------------------------
// Attention + aggregation: block per (b,i). Softmax over j per head in LDS,
// then out[b,i,h,d] = sum_j att * (em[b,i,j,h,d] + xp[b,j,h,d]) + b_node.
// ---------------------------------------------------------------------------
__global__ __launch_bounds__(256)
void attn_kernel(const float* __restrict__ sA, const float* __restrict__ dA,
                 const float* __restrict__ eA, const unsigned char* __restrict__ mask,
                 const float* __restrict__ em, const float* __restrict__ xp,
                 const float* __restrict__ bnode, float* __restrict__ out)
{
    const int bi = blockIdx.x;          // b*128 + i
    const int b  = bi >> 7;
    __shared__ float att[Nn][Hh];
    __shared__ float sminv[Hh];
    const int tid = threadIdx.x;

    if (tid < Nn) {
        int j = tid;
        unsigned char mk = mask[(size_t)bi * Nn + j];
        #pragma unroll
        for (int h = 0; h < Hh; ++h) {
            float lg = sA[bi * Hh + h] + dA[(b * Nn + j) * Hh + h]
                     + eA[((size_t)bi * Nn + j) * Hh + h];
            lg = lg > 0.f ? lg : 0.2f * lg;             // leaky relu
            att[j][h] = mk ? lg : NEG_INF_F;
        }
    }
    __syncthreads();
    if (tid < Hh) {
        int h = tid;
        float m = -3.4e38f;
        for (int j = 0; j < Nn; ++j) m = fmaxf(m, att[j][h]);
        float s = 0.f;
        for (int j = 0; j < Nn; ++j) { float e = __expf(att[j][h] - m); att[j][h] = e; s += e; }
        sminv[h] = 1.f / s;
    }
    __syncthreads();

    const int h = tid >> 6, d = tid & 63;
    const int col = h * DHh + d;
    float acc = 0.f;
    for (int j = 0; j < Nn; ++j)
        acc += att[j][h] * (em[((size_t)bi * Nn + j) * Dc + col]
                          + xp[(size_t)(b * Nn + j) * Dc + col]);
    out[(size_t)bi * Dc + col] = acc * sminv[h] + bnode[col];
}

// ---------------------------------------------------------------------------
// Generic LayerNorm: out = LN(A (+Badd)) * g + beta, optional relu. DD in {256,768}.
// ---------------------------------------------------------------------------
__global__ __launch_bounds__(256)
void ln_kernel(const float* __restrict__ A, const float* __restrict__ Badd,
               const float* __restrict__ g, const float* __restrict__ beta,
               float* __restrict__ out, int DD, int relu)
{
    const int row = blockIdx.x, tid = threadIdx.x;
    const int npt = DD >> 8;
    const float* a = A + (size_t)row * DD;
    const float* bb = Badd ? Badd + (size_t)row * DD : nullptr;
    float vals[3]; float s = 0.f, sq = 0.f;
    for (int p = 0; p < npt; ++p) {
        int c = tid + (p << 8);
        float v = a[c]; if (bb) v += bb[c];
        vals[p] = v; s += v; sq += v * v;
    }
    __shared__ float r1[256], r2[256];
    r1[tid] = s; r2[tid] = sq; __syncthreads();
    for (int st = 128; st > 0; st >>= 1) {
        if (tid < st) { r1[tid] += r1[tid + st]; r2[tid] += r2[tid + st]; }
        __syncthreads();
    }
    float mean = r1[0] / DD;
    float var  = r2[0] / DD - mean * mean;
    float inv  = rsqrtf(var + 1e-5f);
    for (int p = 0; p < npt; ++p) {
        int c = tid + (p << 8);
        float v = (vals[p] - mean) * inv * g[c] + beta[c];
        if (relu) v = v > 0.f ? v : 0.f;
        out[(size_t)row * DD + c] = v;
    }
}

// Edge-update LN: pre[m,:] += t1[b,j] + t2[b,i]; h = relu(LN(pre)); in-place. DD=768.
__global__ __launch_bounds__(256)
void eu_ln_kernel(float* __restrict__ pre, const float* __restrict__ t1,
                  const float* __restrict__ t2, const float* __restrict__ g,
                  const float* __restrict__ beta, int rowBase)
{
    const int tid = threadIdx.x;
    const int gm  = rowBase + blockIdx.x;
    const int b = gm >> 14, rem = gm & 16383;
    const int i = rem >> 7, j = rem & 127;
    float* p = pre + (size_t)blockIdx.x * D3c;
    const float* a1 = t1 + (size_t)(b * Nn + j) * D3c;
    const float* a2 = t2 + (size_t)(b * Nn + i) * D3c;
    float vals[3]; float s = 0.f, sq = 0.f;
    #pragma unroll
    for (int pp = 0; pp < 3; ++pp) {
        int c = tid + (pp << 8);
        float v = p[c] + a1[c] + a2[c];
        vals[pp] = v; s += v; sq += v * v;
    }
    __shared__ float r1[256], r2[256];
    r1[tid] = s; r2[tid] = sq; __syncthreads();
    for (int st = 128; st > 0; st >>= 1) {
        if (tid < st) { r1[tid] += r1[tid + st]; r2[tid] += r2[tid + st]; }
        __syncthreads();
    }
    float mean = r1[0] / D3c;
    float var  = r2[0] / D3c - mean * mean;
    float inv  = rsqrtf(var + 1e-5f);
    #pragma unroll
    for (int pp = 0; pp < 3; ++pp) {
        int c = tid + (pp << 8);
        float v = (vals[pp] - mean) * inv * g[c] + beta[c];
        p[c] = v > 0.f ? v : 0.f;
    }
}

__global__ __launch_bounds__(256)
void copy_kernel(const float* __restrict__ src, float* __restrict__ dst, size_t n)
{
    size_t i = (size_t)blockIdx.x * 256 + threadIdx.x;
    if (i < n) dst[i] = src[i];
}

// ---------------------------------------------------------------------------
extern "C" void kernel_launch(void* const* d_in, const int* in_sizes, int n_in,
                              void* d_out, int out_size, void* d_ws, size_t ws_size,
                              hipStream_t stream)
{
    (void)in_sizes; (void)n_in; (void)out_size; (void)ws_size;
    const float* node_in = (const float*)d_in[0];
    const float* edge_in = (const float*)d_in[1];
    const unsigned char* mask = (const unsigned char*)d_in[2];
    const float* W_V = (const float*)d_in[3];
    const float* W_E = (const float*)d_in[4];
    const float* W_M = (const float*)d_in[5];
    const float* b_M = (const float*)d_in[6];
    const float* att_src  = (const float*)d_in[7];
    const float* att_dst  = (const float*)d_in[8];
    const float* att_edge = (const float*)d_in[9];
    const float* bias_node = (const float*)d_in[10];
    const float* W_ff1 = (const float*)d_in[11];
    const float* b_ff1 = (const float*)d_in[12];
    const float* W_ff2 = (const float*)d_in[13];
    const float* b_ff2 = (const float*)d_in[14];
    const float* ln1_g = (const float*)d_in[15];
    const float* ln1_b = (const float*)d_in[16];
    const float* ln2_g = (const float*)d_in[17];
    const float* ln2_b = (const float*)d_in[18];
    const float* W_eu1a = (const float*)d_in[19];
    const float* W_eu1b = (const float*)d_in[20];
    const float* W_eu1c = (const float*)d_in[21];
    const float* b_eu1  = (const float*)d_in[22];
    const float* eu_ln_g = (const float*)d_in[23];
    const float* eu_ln_b = (const float*)d_in[24];
    const float* W_eu2 = (const float*)d_in[25];
    const float* b_eu2 = (const float*)d_in[26];
    const float* eu_ln1_g = (const float*)d_in[27];
    const float* eu_ln1_b = (const float*)d_in[28];

    // workspace carve-up (floats) ~96 MB
    float* ws = (float*)d_ws;
    size_t o = 0;
    float* nodeB = ws + o; o += (size_t)BNc * Dc;          // 65536
    float* edgeB = ws + o; o += (size_t)BNNc * Dc;         // 8.39M
    float* emB   = ws + o; o += (size_t)BNNc * Dc;         // 8.39M
    float* preB  = ws + o; o += (size_t)CHUNKR * D3c;      // 6.29M
    float* xpB   = ws + o; o += (size_t)BNc * Dc;
    float* outN  = ws + o; o += (size_t)BNc * Dc;
    float* xB    = ws + o; o += (size_t)BNc * Dc;
    float* ff1B  = ws + o; o += (size_t)BNc * Dc;
    float* ff2B  = ws + o; o += (size_t)BNc * Dc;
    float* t1B   = ws + o; o += (size_t)BNc * D3c;
    float* t2B   = ws + o; o += (size_t)BNc * D3c;
    float* sAB   = ws + o; o += (size_t)BNc * Hh;
    float* dAB   = ws + o; o += (size_t)BNc * Hh;
    float* eAB   = ws + o; o += (size_t)BNNc * Hh;
    float* weatt = ws + o; o += (size_t)Dc * Hh;

    auto gemm = [&](const float* A, const float* W, const float* bias,
                    const float* resid, float* C, int M, int K, int Nc, int relu) {
        dim3 g(M / 64, Nc / 64);
        gemm_bf16_wmma<<<g, dim3(128), 0, stream>>>(A, W, bias, resid, C, M, K, Nc, relu);
    };

    const float* nodeSrc = node_in;
    const float* edgeSrc = edge_in;

    for (int l = 0; l < 6; ++l) {
        const size_t wo = (size_t)l * Dc * Dc;
        const size_t ho = (size_t)l * Hh * DHh;   // == l*256
        // --- FCGAT layer ---
        weatt_kernel<<<1, 256, 0, stream>>>(W_E + wo, att_edge + ho, weatt);
        gemm(nodeSrc, W_V + wo, nullptr, nullptr, xpB, BNc, Dc, Dc, 0);
        satt_kernel<<<(BNc * Hh + 255) / 256, 256, 0, stream>>>(
            xpB, att_src + ho, att_dst + ho, sAB, dAB);
        gemm(edgeSrc, W_M + wo, b_M + (size_t)l * Dc, nullptr, emB, BNNc, Dc, Dc, 0);
        eatt_kernel<<<BNNc / 256, 256, 0, stream>>>(edgeSrc, weatt, eAB);
        attn_kernel<<<BNc, 256, 0, stream>>>(sAB, dAB, eAB, mask, emB, xpB,
                                             bias_node + ho, outN);
        ln_kernel<<<BNc, 256, 0, stream>>>(nodeSrc, outN, ln1_g + (size_t)l * Dc,
                                           ln1_b + (size_t)l * Dc, xB, Dc, 0);
        gemm(xB, W_ff1 + wo, b_ff1 + (size_t)l * Dc, nullptr, ff1B, BNc, Dc, Dc, 1);
        gemm(ff1B, W_ff2 + wo, b_ff2 + (size_t)l * Dc, nullptr, ff2B, BNc, Dc, Dc, 0);
        ln_kernel<<<BNc, 256, 0, stream>>>(xB, ff2B, ln2_g + (size_t)l * Dc,
                                           ln2_b + (size_t)l * Dc, nodeB, Dc, 0);
        nodeSrc = nodeB;

        // --- edge update (skip after last conv) ---
        if (l < 5) {
            const size_t wu = (size_t)l * Dc * D3c;   // W_eu1* per-layer
            const size_t w2 = (size_t)l * D3c * Dc;   // W_eu2 per-layer
            gemm(nodeB, W_eu1a + wu, nullptr, nullptr, t1B, BNc, Dc, D3c, 0);
            gemm(nodeB, W_eu1b + wu, nullptr, nullptr, t2B, BNc, Dc, D3c, 0);
            for (int c = 0; c < NCHUNK; ++c) {
                const size_t off = (size_t)c * CHUNKR;
                gemm(edgeSrc + off * Dc, W_eu1c + wu, b_eu1 + (size_t)l * D3c,
                     nullptr, preB, CHUNKR, Dc, D3c, 0);
                eu_ln_kernel<<<CHUNKR, 256, 0, stream>>>(
                    preB, t1B, t2B, eu_ln_g + (size_t)l * D3c,
                    eu_ln_b + (size_t)l * D3c, (int)off);
                gemm(preB, W_eu2 + w2, b_eu2 + (size_t)l * Dc,
                     edgeSrc + off * Dc /*residual*/, emB + off * Dc,
                     CHUNKR, D3c, Dc, 0);
            }
            ln_kernel<<<BNNc, 256, 0, stream>>>(emB, nullptr,
                                                eu_ln1_g + (size_t)l * Dc,
                                                eu_ln1_b + (size_t)l * Dc,
                                                edgeB, Dc, 0);
            edgeSrc = edgeB;
        }
    }

    // outputs: node then edge, concatenated flat
    float* out = (float*)d_out;
    size_t nnode = (size_t)BNc * Dc;
    size_t nedge = (size_t)BNNc * Dc;
    copy_kernel<<<(unsigned)((nnode + 255) / 256), 256, 0, stream>>>(nodeSrc, out, nnode);
    copy_kernel<<<(unsigned)((nedge + 255) / 256), 256, 0, stream>>>(edgeSrc, out + nnode, nedge);
}
